// WindowAttention_3788161155681
// MI455X (gfx1250) — compile-verified, matrix operand order checked by reading, not run on hardware
//
#include <hip/hip_runtime.h>
#include <hip/hip_bf16.h>

// ---------------------------------------------------------------------------
// CDNA5 (gfx1250) fused Swin window-attention, v2:
//   0) one-time f32 -> bf16 conversion of x / qkv_w / proj_w (bf16 x fits L2)
//   1) QKV GEMM  128x64 tiles, K-step 64, bf16 WMMA, async-LDS A staging
//   2) fused attention per (window,head), WMMA 16x16x32 bf16 (hd == K == 32)
//   3) Proj GEMM, same engine, fp32 output
// ---------------------------------------------------------------------------

typedef __attribute__((ext_vector_type(16))) __bf16 v16bf;
typedef __attribute__((ext_vector_type(8)))  __bf16 v8bf;
typedef __attribute__((ext_vector_type(8)))  float  v8f;
typedef __attribute__((ext_vector_type(4)))  int    v4i;

#define NTOK 64
#define HD 32
#define HEADS 16

// GEMM tile config: 128(M) x 64(N), K-step 64, 256 threads (8 waves)
#define GM 128
#define GN 64
#define GK 64
#define GLD (GK + 8)   // padded LDS row (72 elems = 144B, 16B-aligned rows)

#if __has_builtin(__builtin_amdgcn_global_load_async_to_lds_b128)
#define HAVE_ASYNC_LDS 1
#endif

__device__ inline void wait_async0() {
#if defined(HAVE_ASYNC_LDS)
#if __has_builtin(__builtin_amdgcn_s_wait_asynccnt)
  __builtin_amdgcn_s_wait_asynccnt(0);
#else
  asm volatile("s_wait_asynccnt 0x0" ::: "memory");
#endif
#endif
}

// 16-byte global->LDS copy: async (ASYNCcnt-tracked, no VGPR round trip) when
// the toolchain exposes the gfx1250 builtin, plain vector copy otherwise.
// Builtin signature (from hipcc diagnostic): param0 = AS1 v4i* (global src),
// param1 = AS3 v4i* (LDS dst), then imm offset, imm cpol.
__device__ inline void copy16_g2l(__bf16* lds_dst, const __bf16* gsrc) {
#if defined(HAVE_ASYNC_LDS)
  __builtin_amdgcn_global_load_async_to_lds_b128(
      (__attribute__((address_space(1))) v4i*)gsrc,
      (__attribute__((address_space(3))) v4i*)lds_dst, 0, 0);
#else
  *(v8bf*)lds_dst = *(const v8bf*)gsrc;
#endif
}

// Load a 16x32 bf16 A/B fragment (row-major LDS tile, ld elems/row, ld%8==0)
// per the CDNA5 16-bit matrix VGPR layout: lane = half*16 + m; lanes 0-15 hold
// K in {0..7,16..23}, lanes 16-31 hold K in {8..15,24..31}. Two ds_load_b128.
__device__ inline v16bf frag_ld(const __bf16* base, int ld, int row, int kbase,
                                int lane) {
  const __bf16* p = base + row * ld + kbase + ((lane >> 4) << 3);
  v8bf lo = *(const v8bf*)(p);
  v8bf hi = *(const v8bf*)(p + 16);
  v16bf f;
#pragma unroll
  for (int i = 0; i < 8; ++i) { f[i] = lo[i]; f[i + 8] = hi[i]; }
  return f;
}

// ---------------------------------------------------------------------------
// f32 -> bf16 pre-conversion (8 elems/thread, n % 8 == 0)
// ---------------------------------------------------------------------------
__global__ __launch_bounds__(256) void cvt_f32_bf16(
    const float* __restrict__ src, __bf16* __restrict__ dst, int n) {
  int i = (blockIdx.x * 256 + threadIdx.x) * 8;
  if (i >= n) return;
  float4 a = *(const float4*)(src + i);
  float4 b = *(const float4*)(src + i + 4);
  v8bf p;
  p[0] = (__bf16)a.x; p[1] = (__bf16)a.y; p[2] = (__bf16)a.z; p[3] = (__bf16)a.w;
  p[4] = (__bf16)b.x; p[5] = (__bf16)b.y; p[6] = (__bf16)b.z; p[7] = (__bf16)b.w;
  *(v8bf*)(dst + i) = p;
}

// ---------------------------------------------------------------------------
// Tiled GEMM: C[M,N] = A[M,K] @ B[K,N] + bias, all-bf16 inputs, fp32 acc.
// 8 waves: wave w owns subtile row w (16 rows) x all 4 N-subtiles ->
// 8 WMMAs / wave / K-step, A fragment reused 4x.
// ---------------------------------------------------------------------------
template <typename OT>
__global__ __launch_bounds__(256) void gemm_bf16_wmma(
    const __bf16* __restrict__ A, const __bf16* __restrict__ B,
    const float* __restrict__ bias, OT* __restrict__ C,
    int M, int K, int N) {
  __shared__ __align__(16) __bf16 As[GM][GLD];  // 128 x 72 (18 KB)
  __shared__ __align__(16) __bf16 Bt[GN][GLD];  //  64 x 72 ( 9 KB), transposed

  const int tid  = threadIdx.x;
  const int lane = tid & 31;
  const int wave = tid >> 5;          // subtile row 0..7
  const int m0 = blockIdx.x * GM;
  const int n0 = blockIdx.y * GN;
  const int half = lane >> 4;
  const int nc   = lane & 15;

  // staging assignments
  const int ar  = tid >> 1;           // A row 0..127
  const int ac0 = (tid & 1) * 32;     // A col base: 4 x 16B chunks each
  const int bn  = tid & 63;           // B col
  const int bk0 = (tid >> 6) * 16;    // 16 consecutive k per thread

  v8f acc[4] = {{}, {}, {}, {}};

  for (int k0 = 0; k0 < K; k0 += GK) {
    // ---- A tile 128x64 bf16: 4 async 16B copies per thread ----
    const __bf16* ga = A + (size_t)(m0 + ar) * K + (k0 + ac0);
#pragma unroll
    for (int j = 0; j < 4; ++j) copy16_g2l(&As[ar][ac0 + j * 8], ga + j * 8);

    // ---- B tile 64x64 bf16, transposed into LDS (packed b128 stores) ----
    const __bf16* gb = B + (size_t)(k0 + bk0) * N + (n0 + bn);
#pragma unroll
    for (int j = 0; j < 2; ++j) {
      v8bf pk;
#pragma unroll
      for (int e = 0; e < 8; ++e) pk[e] = gb[(size_t)(j * 8 + e) * N];
      *(v8bf*)&Bt[bn][bk0 + j * 8] = pk;
    }
    wait_async0();
    __syncthreads();

#pragma unroll
    for (int kc = 0; kc < 2; ++kc) {
      v16bf a = frag_ld(&As[0][0], GLD, wave * 16 + nc, kc * 32, lane);
#pragma unroll
      for (int ni = 0; ni < 4; ++ni) {
        v16bf b = frag_ld(&Bt[0][0], GLD, ni * 16 + nc, kc * 32, lane);
        acc[ni] = __builtin_amdgcn_wmma_f32_16x16x32_bf16(
            false, a, false, b, (short)0, acc[ni], false, false);
      }
    }
    __syncthreads();
  }

  // ---- epilogue: add bias, store (C layout: lane = n, VGPR r = m) ----
#pragma unroll
  for (int ni = 0; ni < 4; ++ni) {
    int col = n0 + ni * 16 + nc;
    float bs = bias[col];
#pragma unroll
    for (int r = 0; r < 8; ++r) {
      int row = m0 + wave * 16 + r + half * 8;
      C[(size_t)row * N + col] = (OT)(acc[ni][r] + bs);
    }
  }
}

// ---------------------------------------------------------------------------
// Fused window attention. One wave32 per (window b, head h):
//   S(64x64) = 16 WMMA (K=32 == head_dim), softmax in C-layout registers
//   (shfl_xor within 16-lane row groups), O(64x32) = 16 WMMA (K=64).
// Block = 64 threads (2 waves = 2 heads). Grid = (1024 windows, 8).
// ---------------------------------------------------------------------------
__global__ __launch_bounds__(64) void win_attn_wmma(
    const __bf16* __restrict__ qkv,        // [1024*64, 1536] q|k|v packed
    const float* __restrict__ mask,        // [64, 64, 64]
    const float* __restrict__ bias_table,  // [225, 16]
    __bf16* __restrict__ out) {            // [1024*64, 512]  (b,n,h,d)
  __shared__ __align__(16) __bf16 q_s[2][NTOK][HD + 8];
  __shared__ __align__(16) __bf16 k_s[2][NTOK][HD + 8];
  __shared__ __align__(16) __bf16 vt_s[2][HD][NTOK + 8];
  __shared__ __align__(16) __bf16 p_s[2][NTOK][NTOK + 8];

  const int lane = threadIdx.x & 31;
  const int wv   = threadIdx.x >> 5;
  const int b = blockIdx.x;
  const int h = blockIdx.y * 2 + wv;
  const int half = lane >> 4;
  const int nc   = lane & 15;
  const float scale = 0.17677669529663689f;  // 32^-0.5

  const __bf16* qg = qkv + (size_t)b * NTOK * 1536 + h * 32;
  const __bf16* kg = qg + 512;
  const __bf16* vg = qg + 1024;

  for (int i = lane; i < NTOK * HD; i += 32) {
    int t = i >> 5;
    int d = i & 31;
    size_t off = (size_t)t * 1536 + d;
    q_s[wv][t][d]  = (__bf16)((float)qg[off] * scale);
    k_s[wv][t][d]  = kg[off];
    vt_s[wv][d][t] = vg[off];
  }
  __syncthreads();

  const float* maskw = mask + (size_t)(b & 63) * NTOK * NTOK;

  // ---- phase 1: S = q k^T (+bias,+mask), row softmax, P -> LDS bf16 ----
  for (int miT = 0; miT < 4; ++miT) {
    v16bf aq = frag_ld(&q_s[wv][0][0], HD + 8, miT * 16 + nc, 0, lane);
    v8f s[4];
#pragma unroll
    for (int ni = 0; ni < 4; ++ni) {
      v16bf bk = frag_ld(&k_s[wv][0][0], HD + 8, ni * 16 + nc, 0, lane);
      v8f z = {};
      s[ni] = __builtin_amdgcn_wmma_f32_16x16x32_bf16(false, aq, false, bk,
                                                      (short)0, z, false, false);
    }
#pragma unroll
    for (int r = 0; r < 8; ++r) {
      int m = miT * 16 + r + half * 8;
      int mti = m >> 3, mtj = m & 7;
      float rowmax = -3.0e38f;
#pragma unroll
      for (int ni = 0; ni < 4; ++ni) {
        int n = ni * 16 + nc;
        int ri = (mti - (n >> 3) + 7) * 15 + (mtj - (n & 7) + 7);
        float v = s[ni][r] + bias_table[ri * HEADS + h] + maskw[m * NTOK + n];
        s[ni][r] = v;
        rowmax = fmaxf(rowmax, v);
      }
#pragma unroll
      for (int o = 8; o >= 1; o >>= 1)
        rowmax = fmaxf(rowmax, __shfl_xor(rowmax, o, 32));
      float rsum = 0.f;
#pragma unroll
      for (int ni = 0; ni < 4; ++ni) {
        float e = __expf(s[ni][r] - rowmax);
        s[ni][r] = e;
        rsum += e;
      }
#pragma unroll
      for (int o = 8; o >= 1; o >>= 1)
        rsum += __shfl_xor(rsum, o, 32);
      float inv = 1.0f / rsum;
#pragma unroll
      for (int ni = 0; ni < 4; ++ni)
        p_s[wv][m][ni * 16 + nc] = (__bf16)(s[ni][r] * inv);
    }
  }
  __syncthreads();

  // ---- phase 2: O = P @ V (K=64 -> 2 chained WMMA per 16x16 tile) ----
  for (int miT = 0; miT < 4; ++miT) {
#pragma unroll
    for (int nj = 0; nj < 2; ++nj) {
      v8f o = {};
#pragma unroll
      for (int kc = 0; kc < 2; ++kc) {
        v16bf ap = frag_ld(&p_s[wv][0][0], NTOK + 8, miT * 16 + nc, kc * 32, lane);
        v16bf bv = frag_ld(&vt_s[wv][0][0], NTOK + 8, nj * 16 + nc, kc * 32, lane);
        o = __builtin_amdgcn_wmma_f32_16x16x32_bf16(false, ap, false, bv,
                                                    (short)0, o, false, false);
      }
#pragma unroll
      for (int r = 0; r < 8; ++r) {
        int tok = miT * 16 + r + half * 8;
        int d   = nj * 16 + nc;
        out[((size_t)b * NTOK + tok) * 512 + h * 32 + d] = (__bf16)o[r];
      }
    }
  }
}

// ---------------------------------------------------------------------------
extern "C" void kernel_launch(void* const* d_in, const int* in_sizes, int n_in,
                              void* d_out, int out_size, void* d_ws,
                              size_t ws_size, hipStream_t stream) {
  const float* x          = (const float*)d_in[0];  // [1024, 64, 512]
  const float* mask       = (const float*)d_in[1];  // [64, 64, 64]
  const float* qkv_w      = (const float*)d_in[2];  // [512, 1536]
  const float* qkv_b      = (const float*)d_in[3];  // [1536]
  const float* proj_w     = (const float*)d_in[4];  // [512, 512]
  const float* proj_b     = (const float*)d_in[5];  // [512]
  const float* bias_table = (const float*)d_in[6];  // [225, 16]
  float* out = (float*)d_out;

  const int Mrows = 1024 * 64;                   // 65536
  const size_t nX  = (size_t)Mrows * 512;        // 33.5M
  const size_t nWq = 512 * 1536;
  const size_t nWp = 512 * 512;

  __bf16* xbf   = (__bf16*)d_ws;                 // 67 MB (L2-resident)
  __bf16* wqkv  = xbf + nX;                      // 1.5 MB
  __bf16* wproj = wqkv + nWq;                    // 0.5 MB
  __bf16* qkvb  = wproj + nWp;                   // [65536,1536] bf16, 201 MB
  __bf16* attb  = qkvb + (size_t)Mrows * 1536;   // [65536, 512] bf16, 67 MB

  // 0) one-time bf16 down-conversion
  cvt_f32_bf16<<<dim3((nX / 8 + 255) / 256), 256, 0, stream>>>(x, xbf, (int)nX);
  cvt_f32_bf16<<<dim3((nWq / 8 + 255) / 256), 256, 0, stream>>>(qkv_w, wqkv, (int)nWq);
  cvt_f32_bf16<<<dim3((nWp / 8 + 255) / 256), 256, 0, stream>>>(proj_w, wproj, (int)nWp);

  // 1) QKV projection: 103 GFLOP on WMMA bf16
  dim3 g1(Mrows / GM, 1536 / GN);
  gemm_bf16_wmma<__bf16><<<g1, 256, 0, stream>>>(xbf, wqkv, qkv_b, qkvb,
                                                 Mrows, 512, 1536);

  // 2) fused window attention (one wave per (window, head))
  dim3 g2(1024, HEADS / 2);
  win_attn_wmma<<<g2, 64, 0, stream>>>(qkvb, mask, bias_table, attb);

  // 3) output projection (fp32 out)
  dim3 g3(Mrows / GM, 512 / GN);
  gemm_bf16_wmma<float><<<g3, 256, 0, stream>>>(attb, wproj, proj_b, out,
                                                Mrows, 512, 512);
}
